// AdaptiveAttention_59983513256130
// MI455X (gfx1250) — matrix operand order repaired and checked by reading, not compiled, not run on hardware
//
#include <hip/hip_runtime.h>

// ---------------------------------------------------------------------------
// AdaptiveAttention for MI455X (gfx1250, wave32, WMMA f16->f32)
// B=1024, P=196, H=A=512.
//  - spatial[b] parked in LDS as f16 (196KB of 320KB WGP pool): HBM reads the
//    411MB tensor exactly once (~18us floor @ 23.3TB/s).
//  - W_va pre-transposed/converted to f16 [n][k] once: B-operand = two
//    contiguous global_load_b128, no converts in the hot loop.
//  - 105-GFLOP GEMM on v_wmma_f32_16x16x32_f16; logits reduced with shfl_xor
//    + conflict-free ds_add_f32; branchless tanh (v_exp_f32+v_rcp_f32) since
//    tanh runs ~103M times.
//  - Final (context+hidden)@W_ch projection done as a WMMA GEMM, not a GEMV.
// ---------------------------------------------------------------------------

typedef __attribute__((ext_vector_type(16))) _Float16 v16h;
typedef __attribute__((ext_vector_type(8)))  _Float16 v8h;
typedef __attribute__((ext_vector_type(2)))  _Float16 v2h;
typedef __attribute__((ext_vector_type(8)))  float    v8f;

#define B_   1024
#define P_   196
#define H_   512
#define A_   512
#define PPAD 208    // 13 * 16 (M padded to WMMA tiles)
#define ROWF 520    // f16 row stride in LDS (1040 B -> bank rotation)

// Branchless tanh: 1 - 2/(e^{2x}+1). Saturates exactly (+/-1) for large |x|.
__device__ __forceinline__ float fast_tanh(float x) {
  const float e = __expf(2.0f * x);
  return 1.0f - 2.0f * __builtin_amdgcn_rcpf(e + 1.0f);
}

__device__ __forceinline__ v16h cat8h(v8h lo, v8h hi) {
  return __builtin_shufflevector(lo, hi, 0,1,2,3,4,5,6,7,8,9,10,11,12,13,14,15);
}

// ---------------------------------------------------------------------------
// One-time prep: WT[n][k] = (f16)W[k][n].  32x32 LDS tiles, coalesced.
// ---------------------------------------------------------------------------
__global__ __launch_bounds__(256) void transpose_to_f16(
    const float* __restrict__ W, _Float16* __restrict__ WT, int K, int N)
{
  __shared__ float tile[32][33];
  const int k0 = blockIdx.x * 32;
  const int n0 = blockIdx.y * 32;
  const int tx = threadIdx.x & 31;
  const int ty = threadIdx.x >> 5;          // 8 rows of 32
  for (int i = ty; i < 32; i += 8)
    tile[i][tx] = W[(size_t)(k0 + i) * N + n0 + tx];
  __syncthreads();
  for (int i = ty; i < 32; i += 8)
    WT[(size_t)(n0 + i) * K + k0 + tx] = (_Float16)tile[tx][i];
}

// ---------------------------------------------------------------------------
// Generic C = act(X @ W + bias): one wave per 16x16 tile, K streamed in 32s.
// A-lane layout (f16 16x32): lane<16 -> K in [0,16); lane>=16 ->
// [16,32). B-lane layout: lane<16 -> K in [0,16), lane>=16 -> [16,32).
// ACT: 0 = none, 1 = relu, 2 = tanh (compile-time, branchless epilogue).
// ---------------------------------------------------------------------------
template<int ACT>
__global__ __launch_bounds__(256) void gemm16_act(
    const float* __restrict__ X, const float* __restrict__ W,
    const float* __restrict__ bias, float* __restrict__ C,
    int M, int K, int N)
{
  const int wave = threadIdx.x >> 5;
  const int lane = threadIdx.x & 31;
  const int ln   = lane & 15;
  const int hi   = lane >> 4;
  const int tile = blockIdx.x * 8 + wave;
  const int ntn  = N >> 4;
  const int mi   = tile / ntn;
  const int ni   = tile - mi * ntn;
  if (mi * 16 >= M) return;
  const int m = mi * 16 + ln;   // A row for this lane
  const int n = ni * 16 + ln;   // B/D column for this lane

  v8f acc = {};
  for (int kb = 0; kb < K; kb += 32) {
    const int ka  = kb + hi * 16;
    const int kbb = kb + hi * 16;
    v16h a, bm;
#pragma unroll
    for (int j = 0; j < 8; ++j) {
      a[j]     = (_Float16)X[(size_t)m * K + ka + j];
      a[j + 8] = (_Float16)X[(size_t)m * K + ka + 8 + j];
    }
#pragma unroll
    for (int j = 0; j < 16; ++j)
      bm[j] = (_Float16)W[(size_t)(kbb + j) * N + n];
    acc = __builtin_amdgcn_wmma_f32_16x16x32_f16(
        false, a, false, bm, (short)0, acc, false, false);
  }
  const float bn = bias[n];
#pragma unroll
  for (int r = 0; r < 8; ++r) {
    const int row = mi * 16 + r + 8 * hi;   // D: M = r + 8*(lane>=16)
    float v = acc[r] + bn;
    if (ACT == 1)      v = fmaxf(v, 0.0f);
    else if (ACT == 2) v = fast_tanh(v);
    C[(size_t)row * N + n] = v;
  }
}

// ---------------------------------------------------------------------------
// Fused per-batch kernel: WMMA GEMM -> tanh -> dot W_al -> softmax ->
// weighted context from LDS -> +hidden_affine -> vctx (global ws).
// One 256-thread (8-wave) block per batch element; ~219KB dynamic LDS.
// ---------------------------------------------------------------------------
__global__ __launch_bounds__(256) void adaptive_attn_fused(
    const float* __restrict__ spatial, const _Float16* __restrict__ W_vaT,
    const float* __restrict__ b_va,    const float* __restrict__ W_al,
    const float* __restrict__ saff,    const float* __restrict__ haff,
    const float* __restrict__ satt,    const float* __restrict__ hatt,
    float* __restrict__ vctx, float* __restrict__ w_out,
    float* __restrict__ beta_out)
{
  extern __shared__ char smem[];
  _Float16* spat    = (_Float16*)smem;                       // PPAD*ROWF f16
  float* alpha_sm   = (float*)(smem + PPAD * ROWF * 2);      // 208
  float* wts_sm     = alpha_sm + 208;                        // 208
  float* red        = wts_sm + 208;                          // 256

  const int b    = blockIdx.x;
  const int tid  = threadIdx.x;
  const int lane = tid & 31;
  const int wv   = tid >> 5;
  const int ln   = lane & 15;
  const int hi   = lane >> 4;

  // ---- Phase 0: zero logits; spatial[b] f32 -> f16 LDS; zero pad rows ----
  if (tid < P_) alpha_sm[tid] = 0.0f;
  const float4* src = (const float4*)(spatial + (size_t)b * P_ * H_);
  for (int idx = tid; idx < P_ * (H_ / 4); idx += 256) {
    const int p = idx >> 7;        // H_/4 == 128
    const int c = idx & 127;
    float4 v = src[idx];
    _Float16* d = &spat[p * ROWF + c * 4];
    d[0] = (_Float16)v.x; d[1] = (_Float16)v.y;
    d[2] = (_Float16)v.z; d[3] = (_Float16)v.w;
  }
  for (int idx = tid; idx < (PPAD - P_) * ROWF; idx += 256)
    spat[P_ * ROWF + idx] = (_Float16)0.0f;
  __syncthreads();

  // ---- Phase 1: VA = spat @ W_va (WMMA), fused tanh/W_al logit reduce ----
  // Wave handles 4 N-tiles; 13 M-tile accumulators per N-tile so W_vaT is
  // streamed from L2 exactly once per block (2 x b128 per K-step, no cvt).
  for (int ni = wv; ni < (A_ / 16); ni += 8) {
    const int n = ni * 16 + ln;
    const float hatt_n = hatt[(size_t)b * A_ + n];
    const float bva_n  = b_va[n];
    const float wal_n  = W_al[n];
    const _Float16* brow = W_vaT + (size_t)n * H_ + hi * 16;
    v8f acc[13] = {};
    for (int kb = 0; kb < H_; kb += 32) {
      const v8h b0 = *(const v8h*)(brow + kb);       // K: kbb .. kbb+7
      const v8h b1 = *(const v8h*)(brow + kb + 8);   // K: kbb+8 .. kbb+15
      const v16h bm = cat8h(b0, b1);
      const int ka = kb + hi * 16;
#pragma unroll
      for (int mi = 0; mi < 13; ++mi) {
        const _Float16* rp = &spat[(mi * 16 + ln) * ROWF + ka];
        const v8h lo  = *(const v8h*)rp;          // K: ka .. ka+7
        const v8h hi8 = *(const v8h*)(rp + 8);    // K: ka+8 .. ka+15
        const v16h a = cat8h(lo, hi8);
        acc[mi] = __builtin_amdgcn_wmma_f32_16x16x32_f16(
            false, a, false, bm, (short)0, acc[mi], false, false);
      }
    }
    // Reduce 16 column-lanes per row with shfl_xor (masks stay inside the
    // half-wave group), then one conflict-free ds_add_f32 per row.
#pragma unroll
    for (int mi = 0; mi < 13; ++mi) {
#pragma unroll
      for (int r = 0; r < 8; ++r) {
        float t = fast_tanh(acc[mi][r] + hatt_n + bva_n) * wal_n;
        t += __shfl_xor(t, 1);
        t += __shfl_xor(t, 2);
        t += __shfl_xor(t, 4);
        t += __shfl_xor(t, 8);
        const int m = mi * 16 + r + 8 * hi;   // D row mapping
        if (ln == 0 && m < P_) atomicAdd(&alpha_sm[m], t);
      }
    }
  }
  __syncthreads();

  // ---- Phase 2: sentinel logit = sum_a tanh(satt+hatt) * W_al ----
  float sp = 0.0f;
  for (int a2 = tid; a2 < A_; a2 += 256)
    sp += fast_tanh(satt[(size_t)b * A_ + a2] + hatt[(size_t)b * A_ + a2]) * W_al[a2];
  red[tid] = sp;
  __syncthreads();
  for (int s = 128; s > 0; s >>= 1) {
    if (tid < s) red[tid] += red[tid + s];
    __syncthreads();
  }
  if (tid == 0) alpha_sm[P_] = red[0];
  __syncthreads();

  // ---- Phase 3: softmax over 197 logits (b_al is shift-invariant) ----
  const float aval = (tid <= P_) ? alpha_sm[tid] : -3.0e38f;
  red[tid] = aval;
  __syncthreads();
  for (int s = 128; s > 0; s >>= 1) {
    if (tid < s) red[tid] = fmaxf(red[tid], red[tid + s]);
    __syncthreads();
  }
  const float mx = red[0];
  __syncthreads();
  const float e = (tid <= P_) ? __expf(aval - mx) : 0.0f;
  red[tid] = e;
  __syncthreads();
  for (int s = 128; s > 0; s >>= 1) {
    if (tid < s) red[tid] += red[tid + s];
    __syncthreads();
  }
  const float inv = 1.0f / red[0];
  if (tid <= P_) {
    const float wt = e * inv;
    wts_sm[tid] = wt;
    w_out[(size_t)b * (P_ + 1) + tid] = wt;
    if (tid == P_) beta_out[b] = wt;
  }
  __syncthreads();

  // ---- Phase 4: context + hidden_affine -> vctx (consumed by WMMA GEMM) ----
  // Each thread owns a consecutive h-pair: paired-f16 ds_load_b32 reads.
  {
    const int h0 = tid * 2;             // 256 threads * 2 = 512 = H_
    float c0 = saff[(size_t)b * H_ + h0]     * wts_sm[P_];
    float c1 = saff[(size_t)b * H_ + h0 + 1] * wts_sm[P_];
    for (int p = 0; p < P_; ++p) {
      const v2h v = *(const v2h*)&spat[p * ROWF + h0];
      const float w = wts_sm[p];
      c0 += (float)v[0] * w;
      c1 += (float)v[1] * w;
    }
    vctx[(size_t)b * H_ + h0]     = c0 + haff[(size_t)b * H_ + h0];
    vctx[(size_t)b * H_ + h0 + 1] = c1 + haff[(size_t)b * H_ + h0 + 1];
  }
}

// ---------------------------------------------------------------------------
extern "C" void kernel_launch(void* const* d_in, const int* in_sizes, int n_in,
                              void* d_out, int out_size, void* d_ws, size_t ws_size,
                              hipStream_t stream) {
  (void)in_sizes; (void)n_in; (void)out_size; (void)ws_size;
  const float* spatial = (const float*)d_in[0];
  const float* dec     = (const float*)d_in[1];
  const float* st      = (const float*)d_in[2];
  const float* W_sa    = (const float*)d_in[3];
  const float* b_sa    = (const float*)d_in[4];
  const float* W_satt  = (const float*)d_in[5];
  const float* b_satt  = (const float*)d_in[6];
  const float* W_ha    = (const float*)d_in[7];
  const float* b_ha    = (const float*)d_in[8];
  const float* W_hatt  = (const float*)d_in[9];
  const float* b_hatt  = (const float*)d_in[10];
  const float* W_va    = (const float*)d_in[11];
  const float* b_va    = (const float*)d_in[12];
  const float* W_al    = (const float*)d_in[13];
  // d_in[14] = b_al: added uniformly to all logits -> softmax-invariant.
  const float* W_ch    = (const float*)d_in[15];
  const float* b_ch    = (const float*)d_in[16];

  float* ws   = (float*)d_ws;
  float* saff = ws;                               // B*H
  float* haff = ws + 1 * (size_t)B_ * H_;         // B*H
  float* satt = ws + 2 * (size_t)B_ * H_;         // B*H
  float* hatt = ws + 3 * (size_t)B_ * H_;         // B*H
  float* vctx = ws + 4 * (size_t)B_ * H_;         // B*H
  _Float16* W_vaT = (_Float16*)(ws + 5 * (size_t)B_ * H_);  // A*H f16 (512KB)

  float* out      = (float*)d_out;
  float* w_out    = out + (size_t)B_ * H_;
  float* beta_out = w_out + (size_t)B_ * (P_ + 1);

  // One-time weight prep: W_vaT[n][k] = f16(W_va[k][n]).
  transpose_to_f16<<<dim3(H_ / 32, A_ / 32), 256, 0, stream>>>(W_va, W_vaT, H_, A_);

  const int tiles  = (B_ / 16) * (H_ / 16);   // 2048 tiles
  const int blocks = tiles / 8;               // 8 waves per block
  gemm16_act<1><<<blocks, 256, 0, stream>>>(st,   W_sa,   b_sa,   saff, B_, H_, H_);
  gemm16_act<2><<<blocks, 256, 0, stream>>>(dec,  W_ha,   b_ha,   haff, B_, H_, H_);
  gemm16_act<0><<<blocks, 256, 0, stream>>>(saff, W_satt, b_satt, satt, B_, H_, A_);
  gemm16_act<0><<<blocks, 256, 0, stream>>>(haff, W_hatt, b_hatt, hatt, B_, H_, A_);

  const size_t smem = (size_t)PPAD * ROWF * 2 +
                      (208 + 208 + 256) * sizeof(float);  // ~219 KB
  adaptive_attn_fused<<<B_, 256, smem, stream>>>(
      spatial, W_vaT, b_va, W_al,
      saff, haff, satt, hatt, vctx, w_out, beta_out);

  // Final projection as a WMMA GEMM: out = tanh(vctx @ W_ch + b_ch).
  gemm16_act<2><<<blocks, 256, 0, stream>>>(vctx, W_ch, b_ch, out, B_, H_, H_);
}
